// SentenceFour_3478923509978
// MI455X (gfx1250) — compile-verified, matrix operand order checked by reading, not run on hardware
//
#include <hip/hip_runtime.h>

// ---------------------------------------------------------------------------
// SentenceFour transformer stack for gfx1250 (MI455X, wave32, WMMA).
// GEMMs: bf16 WMMA (v_wmma_f32_16x16x32_bf16), f32 accumulate, f32->bf16
// conversion while staging tiles into LDS via float4 loads (no divergent
// guards: K,N are always tile-multiples here; M handled by clamp+select).
// ---------------------------------------------------------------------------

#define D_MODEL 768
#define NHEAD   16
#define HDK     48
#define FFDIM   3072
#define BATCH   4

typedef __bf16 bf16_t;
typedef __attribute__((ext_vector_type(16))) __bf16 v16bf;
typedef __attribute__((ext_vector_type(8)))  __bf16 v8bf;
typedef __attribute__((ext_vector_type(8)))  float  v8f;

__device__ __forceinline__ bf16_t f2bf(float f) {
  union { float f; unsigned u; } x; x.f = f;
  unsigned r = x.u + 0x7FFFu + ((x.u >> 16) & 1u);   // round-to-nearest-even
  union { unsigned short s; bf16_t b; } y;
  y.s = (unsigned short)(r >> 16);
  return y.b;
}

// pack two f32 -> packed bf16x2 dword (lo in [15:0], hi in [31:16])
__device__ __forceinline__ unsigned pack_bf2(float a, float b) {
  union { float f; unsigned u; } x, y; x.f = a; y.f = b;
  unsigned ra = (x.u + 0x7FFFu + ((x.u >> 16) & 1u)) >> 16;
  unsigned rb = (y.u + 0x7FFFu + ((y.u >> 16) & 1u)) & 0xFFFF0000u;
  return ra | rb;
}

// ------------------------- WMMA GEMM: C = act(A @ W + bias) -----------------
// A: MxK f32 row-major, W: KxN f32 row-major, C: MxN f32.
// REQUIRES: K % 32 == 0, N % 128 == 0 (true for every GEMM in this model).
// Workgroup tile 128x128, 8 waves (4 in M x 2 in N), each wave 32x64 (2x4
// 16x16 accumulators). K staged in 32-wide bf16 tiles in LDS (stride 40
// halfwords -> bank-conflict-free 16B ds_load_b128 fragment loads).
#define TM 128
#define TN 128
#define TK 32
#define LDP 40

__global__ __launch_bounds__(256)
void wmma_gemm_bias_act(const float* __restrict__ A, const float* __restrict__ W,
                        const float* __restrict__ bias, float* __restrict__ C,
                        int M, int N, int K, int act) {
  __shared__ __align__(16) bf16_t Asm[TM * LDP];
  __shared__ __align__(16) bf16_t Bsm[TN * LDP];   // stored transposed: [n][k]

  const int tid  = threadIdx.x;
  const int lane = tid & 31;
  const int wave = tid >> 5;
  const int wy = wave >> 1;              // 0..3 -> M direction
  const int wx = wave & 1;               // 0..1 -> N direction
  const int m0 = blockIdx.y * TM;
  const int n0 = blockIdx.x * TN;

  const int l16 = lane & 15;             // row (A) / col (B) within 16x16 tile
  const int hb  = (lane >> 4) << 3;      // K half-select: 0 or 8

  // A-tile loader mapping: 128 rows x 8 float4; 4 vec-loads per thread.
  const int ar  = (tid + 0 * 256) >> 3;          // varies with t below
  (void)ar;

  v8f acc[2][4] = {};

  for (int k0 = 0; k0 < K; k0 += TK) {
    // ---- stage A tile (TM x TK) as float4, f32 -> packed bf16x2 pairs ----
#pragma unroll
    for (int t = 0; t < 4; ++t) {
      int idx = tid + t * 256;                   // 0..1023
      int r   = idx >> 3;                        // 0..127
      int c4  = (idx & 7) << 2;                  // 0,4,...,28
      int gr  = m0 + r;
      int cr  = gr < M ? gr : (M - 1);           // clamp: unconditional load
      float4 vv = *(const float4*)&A[(long)cr * K + (k0 + c4)];
      if (gr >= M) { vv.x = 0.f; vv.y = 0.f; vv.z = 0.f; vv.w = 0.f; }
      uint2 pk; pk.x = pack_bf2(vv.x, vv.y); pk.y = pack_bf2(vv.z, vv.w);
      *reinterpret_cast<uint2*>(&Asm[r * LDP + c4]) = pk;   // 8B aligned
    }
    // ---- stage B tile (TK x TN) as float4, transposed into [n][k] ----
#pragma unroll
    for (int t = 0; t < 4; ++t) {
      int idx = tid + t * 256;                   // 0..1023
      int kk  = idx >> 5;                        // 0..31
      int n4  = (idx & 31) << 2;                 // 0,4,...,124
      float4 vv = *(const float4*)&W[(long)(k0 + kk) * N + (n0 + n4)];
      Bsm[(n4 + 0) * LDP + kk] = f2bf(vv.x);
      Bsm[(n4 + 1) * LDP + kk] = f2bf(vv.y);
      Bsm[(n4 + 2) * LDP + kk] = f2bf(vv.z);
      Bsm[(n4 + 3) * LDP + kk] = f2bf(vv.w);
    }
    // prefetch next K-tile into cache (global_prefetch_b8)
    if (k0 + TK < K) {
      int r  = tid >> 3;
      int gr = m0 + r; int cr = gr < M ? gr : (M - 1);
      __builtin_prefetch(&A[(long)cr * K + (k0 + TK) + ((tid & 7) << 2)], 0, 1);
      __builtin_prefetch(&W[(long)(k0 + TK + (tid >> 5)) * N + (n0 + ((tid & 31) << 2))], 0, 1);
    }
    __syncthreads();

    // ---- fragment loads per ISA 16-bit A/B layout ----
    v16bf afrag[2], bfrag[4];
#pragma unroll
    for (int mi = 0; mi < 2; ++mi) {
      int row = wy * 32 + mi * 16 + l16;
      v8bf lo = *(const v8bf*)&Asm[row * LDP + hb];
      v8bf hi = *(const v8bf*)&Asm[row * LDP + 16 + hb];
#pragma unroll
      for (int e = 0; e < 8; ++e) { afrag[mi][e] = lo[e]; afrag[mi][e + 8] = hi[e]; }
    }
#pragma unroll
    for (int ni = 0; ni < 4; ++ni) {
      int col = wx * 64 + ni * 16 + l16;
      v8bf lo = *(const v8bf*)&Bsm[col * LDP + hb];
      v8bf hi = *(const v8bf*)&Bsm[col * LDP + 16 + hb];
#pragma unroll
      for (int e = 0; e < 8; ++e) { bfrag[ni][e] = lo[e]; bfrag[ni][e + 8] = hi[e]; }
    }

#pragma unroll
    for (int mi = 0; mi < 2; ++mi)
#pragma unroll
      for (int ni = 0; ni < 4; ++ni)
        acc[mi][ni] = __builtin_amdgcn_wmma_f32_16x16x32_bf16(
            false, afrag[mi], false, bfrag[ni], (short)0, acc[mi][ni], false, false);
    __syncthreads();
  }

  // ---- epilogue: C/D layout lane l -> col l%16, rows (l>=16 ? 8 : 0)+r ----
  const int r0 = (lane >> 4) << 3;
#pragma unroll
  for (int mi = 0; mi < 2; ++mi) {
#pragma unroll
    for (int ni = 0; ni < 4; ++ni) {
      int gcol = n0 + wx * 64 + ni * 16 + l16;     // always < N (N%128==0)
      float bv = bias ? bias[gcol] : 0.0f;
#pragma unroll
      for (int r = 0; r < 8; ++r) {
        int grow = m0 + wy * 32 + mi * 16 + r0 + r;
        if (grow < M) {
          float v = acc[mi][ni][r] + bv;
          if (act) v = v > 0.0f ? v : 0.0f;
          C[(long)grow * N + gcol] = v;
        }
      }
    }
  }
}

// ------------------------- LayerNorm (Bessel std, std+eps quirk) ------------
__global__ __launch_bounds__(256)
void layernorm_kernel(const float* __restrict__ x, const float* __restrict__ g,
                      const float* __restrict__ b, float* __restrict__ y, int M) {
  __shared__ float red[256];
  __shared__ float s_mean, s_inv;
  int row = blockIdx.x;
  if (row >= M) return;
  const float* xr = x + (long)row * D_MODEL;
  float s = 0.0f;
  for (int c = threadIdx.x; c < D_MODEL; c += 256) s += xr[c];
  red[threadIdx.x] = s; __syncthreads();
  for (int off = 128; off > 0; off >>= 1) {
    if (threadIdx.x < off) red[threadIdx.x] += red[threadIdx.x + off];
    __syncthreads();
  }
  if (threadIdx.x == 0) s_mean = red[0] / (float)D_MODEL;
  __syncthreads();
  float mean = s_mean, s2 = 0.0f;
  for (int c = threadIdx.x; c < D_MODEL; c += 256) { float d = xr[c] - mean; s2 += d * d; }
  red[threadIdx.x] = s2; __syncthreads();
  for (int off = 128; off > 0; off >>= 1) {
    if (threadIdx.x < off) red[threadIdx.x] += red[threadIdx.x + off];
    __syncthreads();
  }
  if (threadIdx.x == 0) {
    float stdv = sqrtf(red[0] / (float)(D_MODEL - 1));  // unbiased
    s_inv = 1.0f / (stdv + 1e-6f);                       // torch (std+eps) quirk
  }
  __syncthreads();
  float inv = s_inv;
  float* yr = y + (long)row * D_MODEL;
  for (int c = threadIdx.x; c < D_MODEL; c += 256)
    yr[c] = g[c] * (xr[c] - mean) * inv + b[c];
}

// ------------------------- Attention (per (b,h,qi) block) -------------------
__global__ __launch_bounds__(64)
void attention_kernel(const float* __restrict__ q, const float* __restrict__ k,
                      const float* __restrict__ v, float* __restrict__ out,
                      int S, int maskw) {
  __shared__ float qrow[HDK];
  __shared__ float sc[160];
  __shared__ float red[64];
  const int qi = blockIdx.x;
  const int bh = blockIdx.y;
  const int b = bh / NHEAD, h = bh % NHEAD;
  const int tid = threadIdx.x;
  const long rq = ((long)(b * S + qi)) * D_MODEL + h * HDK;
  if (tid < HDK) qrow[tid] = q[rq + tid];
  __syncthreads();

  const float scale = 0.14433756729740643f;  // 1/sqrt(48)
  for (int kk = tid; kk < S; kk += 64) {
    long rk = ((long)(b * S + kk)) * D_MODEL + h * HDK;
    float dot = 0.0f;
#pragma unroll
    for (int d = 0; d < HDK; ++d) dot += qrow[d] * k[rk + d];
    float s = dot * scale;
    if (maskw > 0 && qi < 100 && kk < 100) {          // band only in 100x100 block
      int dd = kk - qi; if (dd < 0) dd = -dd;
      if (dd > maskw) s = -1e9f;
    }
    sc[kk] = s;
  }
  __syncthreads();

  float mx = -3.4e38f;
  for (int kk = tid; kk < S; kk += 64) mx = fmaxf(mx, sc[kk]);
  red[tid] = mx; __syncthreads();
  for (int off = 32; off > 0; off >>= 1) {
    if (tid < off) red[tid] = fmaxf(red[tid], red[tid + off]);
    __syncthreads();
  }
  mx = red[0]; __syncthreads();

  float sm = 0.0f;
  for (int kk = tid; kk < S; kk += 64) { float e = __expf(sc[kk] - mx); sc[kk] = e; sm += e; }
  red[tid] = sm; __syncthreads();
  for (int off = 32; off > 0; off >>= 1) {
    if (tid < off) red[tid] += red[tid + off];
    __syncthreads();
  }
  float inv = 1.0f / red[0];
  __syncthreads();

  for (int d = tid; d < HDK; d += 64) {
    float o = 0.0f;
    for (int kk = 0; kk < S; ++kk)
      o += sc[kk] * v[((long)(b * S + kk)) * D_MODEL + h * HDK + d];
    out[rq + d] = o * inv;
  }
}

// ------------------------- small elementwise / assembly kernels -------------
__global__ void add_batch0_kernel(float* __restrict__ x, const float* __restrict__ src, int S) {
  long n = (long)BATCH * S * D_MODEL;
  long i = blockIdx.x * (long)blockDim.x + threadIdx.x;
  if (i >= n) return;
  x[i] += src[i % ((long)S * D_MODEL)];   // batch-0 slice broadcast quirk
}

__global__ void build_acat_kernel(const float* __restrict__ sent,
                                  const float* __restrict__ gimg, float* __restrict__ A) {
  long n = (long)BATCH * 100 * (2 * D_MODEL);
  long i = blockIdx.x * (long)blockDim.x + threadIdx.x;
  if (i >= n) return;
  int c = (int)(i % (2 * D_MODEL));
  long row = i / (2 * D_MODEL);
  int b = (int)(row / 100);
  A[i] = (c < D_MODEL) ? sent[row * D_MODEL + c] : gimg[(long)b * D_MODEL + (c - D_MODEL)];
}

__global__ void build_gt_kernel(const float* __restrict__ sf, const float* __restrict__ img,
                                const float* __restrict__ se, const float* __restrict__ ee,
                                const float* __restrict__ cls, int enc_idx,
                                float* __restrict__ x) {
  long n = (long)BATCH * 142 * D_MODEL;
  long i = blockIdx.x * (long)blockDim.x + threadIdx.x;
  if (i >= n) return;
  int c = (int)(i % D_MODEL);
  long row = i / D_MODEL;
  int b = (int)(row / 142), t = (int)(row % 142);
  float v;
  if (t < 100)       v = sf[((long)(b * 100 + t)) * D_MODEL + c];
  else if (t < 139)  v = img[((long)(b * 39 + (t - 100))) * D_MODEL + c];
  else if (t == 139) v = se[(long)b * D_MODEL + c];
  else if (t == 140) v = ee[(long)b * D_MODEL + c];
  else               v = cls[(long)enc_idx * D_MODEL + c];
  x[i] = v;
}

__global__ void build_gt5_kernel(const float* __restrict__ sf, const float* __restrict__ ee,
                                 const float* __restrict__ cls, float* __restrict__ x) {
  long n = (long)BATCH * 102 * D_MODEL;
  long i = blockIdx.x * (long)blockDim.x + threadIdx.x;
  if (i >= n) return;
  int c = (int)(i % D_MODEL);
  long row = i / D_MODEL;
  int b = (int)(row / 102), t = (int)(row % 102);
  float v;
  if (t == 0)      v = cls[4L * D_MODEL + c];
  else if (t == 1) v = ee[(long)b * D_MODEL + c];
  else             v = sf[((long)(b * 100 + (t - 2))) * D_MODEL + c];
  x[i] = v;
}

__global__ void extract_cls_kernel(const float* __restrict__ y, float* __restrict__ u22,
                                   int enc_idx) {
  int i = blockIdx.x * blockDim.x + threadIdx.x;
  if (i >= BATCH * D_MODEL) return;
  int b = i / D_MODEL, c = i % D_MODEL;
  u22[((long)b * 4 + enc_idx) * D_MODEL + c] = y[((long)(b * 142 + 141)) * D_MODEL + c];
}

__global__ void copy_y5_kernel(const float* __restrict__ y5, float* __restrict__ out) {
  int i = blockIdx.x * blockDim.x + threadIdx.x;
  if (i >= BATCH * D_MODEL) return;
  int b = i / D_MODEL, c = i % D_MODEL;
  out[i] = y5[((long)(b * 102)) * D_MODEL + c];
}

__global__ void s2tsa_final_kernel(const float* __restrict__ f2, const float* __restrict__ u22,
                                   float* __restrict__ out) {
  int i = blockIdx.x * blockDim.x + threadIdx.x;
  if (i >= BATCH * D_MODEL) return;
  int b = i / D_MODEL, c = i % D_MODEL;
  float v[4], e[4];
  float mx = -3.4e38f;
#pragma unroll
  for (int j = 0; j < 4; ++j) { v[j] = f2[((long)b * 4 + j) * D_MODEL + c]; mx = fmaxf(mx, v[j]); }
  float sum = 0.0f;
#pragma unroll
  for (int j = 0; j < 4; ++j) { e[j] = __expf(v[j] - mx); sum += e[j]; }
  float o = 0.0f;
#pragma unroll
  for (int j = 0; j < 4; ++j) o += (e[j] / sum) * u22[((long)b * 4 + j) * D_MODEL + c];
  out[i] = o;
}

// ------------------------- host orchestration -------------------------------
static inline void launch_gemm(const float* A, const float* W, const float* bias, float* C,
                               int M, int N, int K, int act, hipStream_t stream) {
  dim3 grid((N + TN - 1) / TN, (M + TM - 1) / TM);
  wmma_gemm_bias_act<<<grid, dim3(256), 0, stream>>>(A, W, bias, C, M, N, K, act);
}

static void run_encoder(const float* const* p, float* x, int S, int maskw,
                        float* h, float* q, float* k, float* v, float* att,
                        float* proj, float* ffh, float* y, hipStream_t stream) {
  const int M = BATCH * S;
  const long n = (long)M * D_MODEL;
  const int eb = (int)((n + 255) / 256);

  layernorm_kernel<<<M, 256, 0, stream>>>(x, p[12], p[13], h, M);      // ln1
  launch_gemm(h, p[0], p[1], q, M, D_MODEL, D_MODEL, 0, stream);       // q
  launch_gemm(h, p[2], p[3], k, M, D_MODEL, D_MODEL, 0, stream);       // k
  launch_gemm(h, p[4], p[5], v, M, D_MODEL, D_MODEL, 0, stream);       // v
  attention_kernel<<<dim3(S, BATCH * NHEAD), 64, 0, stream>>>(q, k, v, att, S, maskw);
  launch_gemm(att, p[6], p[7], proj, M, D_MODEL, D_MODEL, 0, stream);  // o proj
  add_batch0_kernel<<<eb, 256, 0, stream>>>(x, proj, S);

  layernorm_kernel<<<M, 256, 0, stream>>>(x, p[14], p[15], h, M);      // ln2
  launch_gemm(h, p[8], p[9], ffh, M, FFDIM, D_MODEL, 1, stream);       // ff1 + relu
  launch_gemm(ffh, p[10], p[11], proj, M, D_MODEL, FFDIM, 0, stream);  // ff2
  add_batch0_kernel<<<eb, 256, 0, stream>>>(x, proj, S);

  layernorm_kernel<<<M, 256, 0, stream>>>(x, p[16], p[17], y, M);      // lnf
}

extern "C" void kernel_launch(void* const* d_in, const int* in_sizes, int n_in,
                              void* d_out, int out_size, void* d_ws, size_t ws_size,
                              hipStream_t stream) {
  (void)in_sizes; (void)n_in; (void)out_size; (void)ws_size;

  const float* sent = (const float*)d_in[0];
  const float* img  = (const float*)d_in[1];
  const float* gimg = (const float*)d_in[2];
  const float* ee   = (const float*)d_in[3];
  const float* se   = (const float*)d_in[4];
  const float* f_W3_w   = (const float*)d_in[5];
  const float* f_W3_b   = (const float*)d_in[6];
  const float* s2t_W1_w = (const float*)d_in[7];
  const float* s2t_W1_b = (const float*)d_in[8];
  const float* s2t_W_w  = (const float*)d_in[9];
  const float* s2t_W_b  = (const float*)d_in[10];
  const float* cls      = (const float*)d_in[11];
  const float* encp[5][18];
  for (int e = 0; e < 5; ++e)
    for (int j = 0; j < 18; ++j)
      encp[e][j] = (const float*)d_in[12 + e * 18 + j];

  // ---- workspace carve-out (f32) ----
  float* w = (float*)d_ws;
  size_t off = 0;
  auto alloc = [&](size_t nelem) { float* p = w + off; off += nelem; return p; };
  float* Acat = alloc((size_t)BATCH * 100 * 2 * D_MODEL);  // 400x1536
  float* sf   = alloc((size_t)BATCH * 100 * D_MODEL);      // 400x768
  float* x    = alloc((size_t)BATCH * 142 * D_MODEL);
  float* h    = alloc((size_t)BATCH * 142 * D_MODEL);
  float* qb   = alloc((size_t)BATCH * 142 * D_MODEL);
  float* kb   = alloc((size_t)BATCH * 142 * D_MODEL);
  float* vb   = alloc((size_t)BATCH * 142 * D_MODEL);
  float* att  = alloc((size_t)BATCH * 142 * D_MODEL);
  float* proj = alloc((size_t)BATCH * 142 * D_MODEL);
  float* ffh  = alloc((size_t)BATCH * 142 * FFDIM);
  float* y    = alloc((size_t)BATCH * 142 * D_MODEL);
  float* u22  = alloc((size_t)BATCH * 4 * D_MODEL);
  float* s2f  = alloc((size_t)BATCH * 4 * D_MODEL);
  float* s2f2 = alloc((size_t)BATCH * 4 * D_MODEL);

  // ---- sf = relu([sentence | broadcast(global)] @ f_W3 + b) ----
  {
    long n = (long)BATCH * 100 * 2 * D_MODEL;
    build_acat_kernel<<<(int)((n + 255) / 256), 256, 0, stream>>>(sent, gimg, Acat);
    launch_gemm(Acat, f_W3_w, f_W3_b, sf, BATCH * 100, D_MODEL, 2 * D_MODEL, 1, stream);
  }

  // ---- 4 banded encoders on 142-token sequence ----
  const int maskw[4] = {0, 1, 2, 3};   // 0 = no mask; else band half-width
  for (int i = 0; i < 4; ++i) {
    long n = (long)BATCH * 142 * D_MODEL;
    build_gt_kernel<<<(int)((n + 255) / 256), 256, 0, stream>>>(sf, img, se, ee, cls, i, x);
    run_encoder(encp[i], x, 142, maskw[i], h, qb, kb, vb, att, proj, ffh, y, stream);
    extract_cls_kernel<<<(BATCH * D_MODEL + 255) / 256, 256, 0, stream>>>(y, u22, i);
  }

  // ---- encoder 5 on 102-token sequence ----
  {
    long n = (long)BATCH * 102 * D_MODEL;
    build_gt5_kernel<<<(int)((n + 255) / 256), 256, 0, stream>>>(sf, ee, cls, x);
    run_encoder(encp[4], x, 102, 0, h, qb, kb, vb, att, proj, ffh, y, stream);
    copy_y5_kernel<<<(BATCH * D_MODEL + 255) / 256, 256, 0, stream>>>(
        y, (float*)d_out + BATCH * D_MODEL);
  }

  // ---- s2tsa head ----
  launch_gemm(u22, s2t_W1_w, s2t_W1_b, s2f, BATCH * 4, D_MODEL, D_MODEL, 1, stream);
  launch_gemm(s2f, s2t_W_w, s2t_W_b, s2f2, BATCH * 4, D_MODEL, D_MODEL, 0, stream);
  s2tsa_final_kernel<<<(BATCH * D_MODEL + 255) / 256, 256, 0, stream>>>(
      s2f2, u22, (float*)d_out);
}
